// FusedGPTJAttentionForQuantizedModel_1726576854711
// MI455X (gfx1250) — compile-verified, matrix operand order checked by reading, not run on hardware
//
#include <hip/hip_runtime.h>
#include <hip/hip_bf16.h>

typedef _Float16 half_t;
typedef __attribute__((ext_vector_type(16))) _Float16 v16h;
typedef __attribute__((ext_vector_type(8)))  _Float16 v8h;
typedef __attribute__((ext_vector_type(8)))  float    v8f;

#define B_    2
#define S_    2048
#define D_    4096
#define H_    16
#define HD_   256
#define ROT_  64
#define NQKV  (3 * D_)
#define M_    (B_ * S_)   // 4096 rows

#define WMMA_F16(a, b, c) \
  __builtin_amdgcn_wmma_f32_16x16x32_f16(false, (a), false, (b), (short)0, (c), false, false)

// Load a 16x32 f16 WMMA fragment (A pattern; B pattern is identical when the
// source is stored [N][K] row-major). Per ISA 7.12.2: lane L -> row = L%16,
// halves cover K as two contiguous 8-half chunks at 8*(L>>4) and 16+8*(L>>4).
__device__ __forceinline__ v16h load_frag(const half_t* p0, int stride, int lane) {
  const half_t* p = p0 + (size_t)(lane & 15) * (size_t)stride + ((lane >> 4) << 3);
  v16h f;
  *(v8h*)&f       = *(const v8h*)(p);
  *((v8h*)&f + 1) = *(const v8h*)(p + 16);
  return f;
}

// ---------------------------------------------------------------- f32 -> f16
__global__ void k_f32_to_f16(const float* __restrict__ in, half_t* __restrict__ out, int n) {
  int i = blockIdx.x * blockDim.x + threadIdx.x;
  if (i < n) out[i] = (half_t)in[i];
}

// ------------------------------------------------- GPTQ int4 dequant -> Wt[N][K]
// qweight [K/8][N] int32 (8 x 4-bit along k), qzeros [G][N/8], scales [G][N].
// g_idx[k] == k/128 (GS=128), so g = (kk*8)/128 = kk>>4. One thread per packed
// word -> 8 consecutive k values of column n -> one 16-byte store into Wt[n][k..k+7].
__global__ void k_dequant(const int* __restrict__ qw, const int* __restrict__ qz,
                          const float* __restrict__ sc, half_t* __restrict__ Wt,
                          int N, int K) {
  int idx = blockIdx.x * blockDim.x + threadIdx.x;
  int total = (K >> 3) * N;
  if (idx >= total) return;
  int n  = idx % N;
  int kk = idx / N;
  int w  = qw[(size_t)kk * N + n];
  int g  = kk >> 4;
  int z  = ((qz[(size_t)g * (N >> 3) + (n >> 3)] >> ((n & 7) * 4)) & 15) + 1;
  float s = sc[(size_t)g * N + n];
  v8h o;
#pragma unroll
  for (int i = 0; i < 8; ++i)
    o[i] = (half_t)(s * (float)(((w >> (4 * i)) & 15) - z));
  *(v8h*)(Wt + (size_t)n * K + ((size_t)kk << 3)) = o;
}

// ---------------------------------------------------------------- WMMA GEMM
// C[M][N] = A[M][K] * Bt[N][K]^T.  Block: 256 threads (8 waves), tile 64x128,
// wave tile 32x32 (4 accumulators), k-step 64 (two WMMA k-slices per stage).
template <bool F16OUT>
__global__ __launch_bounds__(256) void k_gemm_wmma(
    const half_t* __restrict__ A, const half_t* __restrict__ Bt,
    void* __restrict__ Cout, int M, int N, int K) {
  __shared__ half_t As[64 * 72];    // 72-half stride: 144B rows, 16B aligned chunks
  __shared__ half_t Bs[128 * 72];

  const int t    = threadIdx.x;
  const int lane = t & 31;
  const int w    = t >> 5;
  const int wm   = (w >> 2) * 32;   // 0 / 32
  const int wn   = (w & 3) * 32;    // 0..96
  const int m0   = blockIdx.y * 64;
  const int n0   = blockIdx.x * 128;

  v8f acc00 = {}, acc01 = {}, acc10 = {}, acc11 = {};

  for (int k0 = 0; k0 < K; k0 += 64) {
    // Stage A tile 64x64 halves (512 16B chunks / 2 per thread)
#pragma unroll
    for (int i = 0; i < 2; ++i) {
      int c = t + i * 256;
      int rr = c >> 3, cc = (c & 7) << 3;
      *(v8h*)&As[rr * 72 + cc] = *(const v8h*)(A + (size_t)(m0 + rr) * K + k0 + cc);
    }
    // Stage B tile 128x64 halves (1024 chunks / 4 per thread)
#pragma unroll
    for (int i = 0; i < 4; ++i) {
      int c = t + i * 256;
      int rr = c >> 3, cc = (c & 7) << 3;
      *(v8h*)&Bs[rr * 72 + cc] = *(const v8h*)(Bt + (size_t)(n0 + rr) * K + k0 + cc);
    }
    // Speculative prefetch of next k-tile (gfx1250 global_prefetch path)
    if (k0 + 64 < K) {
      __builtin_prefetch(A + (size_t)(m0 + (t >> 2)) * K + k0 + 64, 0, 1);
      __builtin_prefetch(Bt + (size_t)(n0 + (t >> 1)) * K + k0 + 64, 0, 1);
    }
    __syncthreads();
#pragma unroll
    for (int kk = 0; kk < 64; kk += 32) {
      v16h a0 = load_frag(&As[(wm + 0)  * 72 + kk], 72, lane);
      v16h a1 = load_frag(&As[(wm + 16) * 72 + kk], 72, lane);
      v16h b0 = load_frag(&Bs[(wn + 0)  * 72 + kk], 72, lane);
      v16h b1 = load_frag(&Bs[(wn + 16) * 72 + kk], 72, lane);
      acc00 = WMMA_F16(a0, b0, acc00);
      acc01 = WMMA_F16(a0, b1, acc01);
      acc10 = WMMA_F16(a1, b0, acc10);
      acc11 = WMMA_F16(a1, b1, acc11);
    }
    __syncthreads();
  }

  // Epilogue: lane L, element r -> m = r + 8*(L>>4), n = L&15 (ISA C/D layout)
  const int mrow = 8 * (lane >> 4);
  const int ncol = lane & 15;
#pragma unroll
  for (int mt = 0; mt < 2; ++mt)
#pragma unroll
    for (int nt = 0; nt < 2; ++nt) {
      const v8f& acc = mt == 0 ? (nt == 0 ? acc00 : acc01)
                               : (nt == 0 ? acc10 : acc11);
#pragma unroll
      for (int r = 0; r < 8; ++r) {
        size_t m = (size_t)(m0 + wm + mt * 16 + r + mrow);
        size_t n = (size_t)(n0 + wn + nt * 16 + ncol);
        if (F16OUT) ((half_t*)Cout)[m * N + n] = (half_t)acc[r];
        else        ((float*)Cout)[m * N + n]  = acc[r];
      }
    }
}

// ------------------------------------------------- RoPE + layout rearrangement
// qkv [B*S][3D] f16 -> Qr,Kr [B,H,S,hd] (Q scaled by 1/sqrt(hd)), Vt [B,H,hd,S].
__global__ void k_rope_layout(const half_t* __restrict__ qkv,
                              half_t* __restrict__ Qr, half_t* __restrict__ Kr,
                              half_t* __restrict__ Vt) {
  int idx = blockIdx.x * blockDim.x + threadIdx.x;   // B*S*H*(hd/2) = 2^23
  int d2 = idx & 127;
  int h  = (idx >> 7) & 15;
  int s  = (idx >> 11) & 2047;
  int b  = idx >> 22;
  if (b >= B_) return;
  int d = d2 << 1;
  const half_t* row = qkv + (size_t)(b * S_ + s) * NQKV;
  int c = h * HD_ + d;
  float q0 = (float)row[c],            q1 = (float)row[c + 1];
  float k0 = (float)row[D_ + c],       k1 = (float)row[D_ + c + 1];
  float v0 = (float)row[2 * D_ + c],   v1 = (float)row[2 * D_ + c + 1];
  if (d < ROT_) {
    float inv = powf(10000.0f, -(float)d * (1.0f / (float)ROT_));
    float ang = (float)s * inv;
    float sn, cs;
    sincosf(ang, &sn, &cs);
    float tq0 = q0 * cs - q1 * sn, tq1 = q1 * cs + q0 * sn;
    float tk0 = k0 * cs - k1 * sn, tk1 = k1 * cs + k0 * sn;
    q0 = tq0; q1 = tq1; k0 = tk0; k1 = tk1;
  }
  const float qs = 0.0625f;   // 1/sqrt(256), folded softmax scale
  size_t qk = ((size_t)(b * H_ + h) * S_ + s) * HD_ + d;
  Qr[qk]     = (half_t)(q0 * qs);
  Qr[qk + 1] = (half_t)(q1 * qs);
  Kr[qk]     = (half_t)k0;
  Kr[qk + 1] = (half_t)k1;
  size_t vt = ((size_t)(b * H_ + h) * HD_ + d) * S_ + s;
  Vt[vt]      = (half_t)v0;
  Vt[vt + S_] = (half_t)v1;
}

// --------------------------------------------------------- flash attention
// Block = (q-tile of 32 rows) x (b,h). 8 waves. Per 128-key tile:
//   each wave computes a 32x16 S slice (K frags straight from global),
//   online softmax in LDS, each wave accumulates a 32-wide d-slice of O
//   (V frags straight from global Vt), rescaled by alpha each step.
__global__ __launch_bounds__(256) void k_attention(
    const half_t* __restrict__ Qr, const half_t* __restrict__ Kr,
    const half_t* __restrict__ Vt, half_t* __restrict__ attn) {
  __shared__ half_t Qs[32 * 264];     // 32 x 256 (+8 pad)
  __shared__ float  Ss[32 * 136];     // 32 x 128 (+8 pad) f32 scores
  __shared__ half_t Ps[32 * 136];     // probabilities (f16)
  __shared__ float  row_max[32], row_sum[32], alpha_s[32];

  const int t    = threadIdx.x;
  const int lane = t & 31;
  const int w    = t >> 5;
  const int qt   = blockIdx.x;
  const int bh   = blockIdx.y;
  const int b    = bh >> 4;
  const int h    = bh & 15;
  const int mrow = 8 * (lane >> 4);
  const int ncl  = lane & 15;

  // Load the 32x256 Q tile into LDS
  const half_t* Qbase = Qr + ((size_t)bh * S_ + (size_t)qt * 32) * HD_;
#pragma unroll
  for (int i = 0; i < 4; ++i) {
    int c = t + i * 256;                 // 1024 chunks of 8 halves
    int rr = c >> 5, cc = (c & 31) << 3;
    *(v8h*)&Qs[rr * 264 + cc] = *(const v8h*)(Qbase + (size_t)rr * HD_ + cc);
  }
  if (t < 32) { row_max[t] = -3.0e38f; row_sum[t] = 0.0f; }
  __syncthreads();

  v8f o00 = {}, o01 = {}, o10 = {}, o11 = {};
  const int qi_max = qt * 32 + 31;
  const int nj = qi_max / 128 + 1;

  for (int j = 0; j < nj; ++j) {
    const int kbase = j * 128;

    // ---- S = Q K^T for this wave's 16-key slice
    v8f s0 = {}, s1 = {};
    const half_t* Kb = Kr + ((size_t)bh * S_ + kbase + w * 16) * HD_;
#pragma unroll
    for (int d0 = 0; d0 < HD_; d0 += 32) {
      v16h a0 = load_frag(&Qs[0  * 264 + d0], 264, lane);
      v16h a1 = load_frag(&Qs[16 * 264 + d0], 264, lane);
      v16h bf = load_frag(Kb + d0, HD_, lane);
      s0 = WMMA_F16(a0, bf, s0);
      s1 = WMMA_F16(a1, bf, s1);
    }
    const int ncol = w * 16 + ncl;
    const int kj   = kbase + ncol;
#pragma unroll
    for (int r = 0; r < 8; ++r) {
      int m = r + mrow;
      Ss[m        * 136 + ncol] = (kj <= qt * 32 + m)      ? s0[r] : -3.0e38f;
      Ss[(m + 16) * 136 + ncol] = (kj <= qt * 32 + m + 16) ? s1[r] : -3.0e38f;
    }
    __syncthreads();

    // ---- online softmax, one thread per q row
    if (t < 32) {
      const int m = t, qi = qt * 32 + m;
      int lim = qi - kbase + 1; if (lim > 128) lim = 128;
      float mx = row_max[m];
      for (int c2 = 0; c2 < lim; ++c2) mx = fmaxf(mx, Ss[m * 136 + c2]);
      float al  = __expf(row_max[m] - mx);
      float sum = row_sum[m] * al;
      for (int c2 = 0; c2 < lim; ++c2) {
        float p = __expf(Ss[m * 136 + c2] - mx);
        Ps[m * 136 + c2] = (half_t)p;
        sum += p;
      }
      for (int c2 = lim; c2 < 128; ++c2) Ps[m * 136 + c2] = (half_t)0.0f;
      row_max[m] = mx; row_sum[m] = sum; alpha_s[m] = al;
    }
    __syncthreads();

    // ---- rescale running accumulators by alpha
#pragma unroll
    for (int r = 0; r < 8; ++r) {
      float a0s = alpha_s[r + mrow];
      float a1s = alpha_s[16 + r + mrow];
      o00[r] *= a0s; o01[r] *= a0s;
      o10[r] *= a1s; o11[r] *= a1s;
    }

    // ---- O += P V  (this wave owns d columns [w*32, w*32+32))
    const half_t* Vb = Vt + ((size_t)bh * HD_ + w * 32) * S_ + kbase;
#pragma unroll
    for (int kk = 0; kk < 128; kk += 32) {
      v16h p0  = load_frag(&Ps[0  * 136 + kk], 136, lane);
      v16h p1  = load_frag(&Ps[16 * 136 + kk], 136, lane);
      v16h v0f = load_frag(Vb + kk, S_, lane);
      v16h v1f = load_frag(Vb + (size_t)16 * S_ + kk, S_, lane);
      o00 = WMMA_F16(p0, v0f, o00);
      o01 = WMMA_F16(p0, v1f, o01);
      o10 = WMMA_F16(p1, v0f, o10);
      o11 = WMMA_F16(p1, v1f, o11);
    }
    __syncthreads();
  }

  // ---- normalize and write attn output as [B, S, H*hd] f16
#pragma unroll
  for (int r = 0; r < 8; ++r) {
    int m0r = r + mrow;
    float inv0 = 1.0f / row_sum[m0r];
    float inv1 = 1.0f / row_sum[m0r + 16];
    int s0r = qt * 32 + m0r;
    size_t base0 = ((size_t)(b * S_ + s0r))      * D_ + h * HD_ + w * 32;
    size_t base1 = ((size_t)(b * S_ + s0r + 16)) * D_ + h * HD_ + w * 32;
    attn[base0 + ncl]      = (half_t)(o00[r] * inv0);
    attn[base0 + 16 + ncl] = (half_t)(o01[r] * inv0);
    attn[base1 + ncl]      = (half_t)(o10[r] * inv1);
    attn[base1 + 16 + ncl] = (half_t)(o11[r] * inv1);
  }
}

// --------------------------------------------------------------------------
extern "C" void kernel_launch(void* const* d_in, const int* in_sizes, int n_in,
                              void* d_out, int out_size, void* d_ws, size_t ws_size,
                              hipStream_t stream) {
  const float* hidden = (const float*)d_in[0];
  const int*   qw_qkv = (const int*)d_in[1];
  const int*   qz_qkv = (const int*)d_in[2];
  const float* sc_qkv = (const float*)d_in[3];
  const int*   qw_out = (const int*)d_in[4];
  const int*   qz_out = (const int*)d_in[5];
  const float* sc_out = (const float*)d_in[6];
  // d_in[7] = g_idx; identically k/GS, computed inline in k_dequant.
  float* out = (float*)d_out;

  char* ws = (char*)d_ws;
  auto take = [&](size_t bytes) -> char* {
    char* p = ws;
    ws += (bytes + 255) & ~(size_t)255;
    return p;
  };
  half_t* A16  = (half_t*)take((size_t)M_ * D_ * 2);          //  32 MB
  half_t* Wqkv = (half_t*)take((size_t)D_ * NQKV * 2);        //  96 MB
  half_t* Wout = (half_t*)take((size_t)D_ * D_ * 2);          //  32 MB
  half_t* qkv  = (half_t*)take((size_t)M_ * NQKV * 2);        //  96 MB
  half_t* Qr   = (half_t*)take((size_t)M_ * D_ * 2);          //  32 MB
  half_t* Kr   = (half_t*)take((size_t)M_ * D_ * 2);          //  32 MB
  half_t* Vtb  = (half_t*)take((size_t)M_ * D_ * 2);          //  32 MB
  half_t* attn = (half_t*)take((size_t)M_ * D_ * 2);          //  32 MB

  k_f32_to_f16<<<(M_ * D_) / 256, 256, 0, stream>>>(hidden, A16, M_ * D_);

  k_dequant<<<((D_ / 8) * NQKV) / 256, 256, 0, stream>>>(qw_qkv, qz_qkv, sc_qkv, Wqkv, NQKV, D_);
  k_dequant<<<((D_ / 8) * D_) / 256, 256, 0, stream>>>(qw_out, qz_out, sc_out, Wout, D_, D_);

  k_gemm_wmma<true><<<dim3(NQKV / 128, M_ / 64), 256, 0, stream>>>(A16, Wqkv, qkv, M_, NQKV, D_);

  k_rope_layout<<<(B_ * S_ * H_ * (HD_ / 2)) / 256, 256, 0, stream>>>(qkv, Qr, Kr, Vtb);

  k_attention<<<dim3(S_ / 32, B_ * H_), 256, 0, stream>>>(Qr, Kr, Vtb, attn);

  k_gemm_wmma<false><<<dim3(D_ / 128, M_ / 64), 256, 0, stream>>>(attn, Wout, out, M_, D_, D_);
}
